// Semantic_Mapping_12000138625128
// MI455X (gfx1250) — matrix operand order repaired and checked by reading, not compiled, not run on hardware
//
#include <hip/hip_runtime.h>
#include <math.h>

typedef __attribute__((ext_vector_type(16))) _Float16 v16h;
typedef __attribute__((ext_vector_type(8)))  float    v8f;

namespace {
constexpr int kFH = 480, kFW = 640, kN = kFH * kFW;       // 307200 pixels
constexpr int kCObs = 20, kNSem = 16;
constexpr int kVR = 100, kHZ = 80;
constexpr int kVoxN = kVR * kVR * kHZ;                    // 800000 cells / channel
constexpr int kCFeat = 17;
constexpr int kMap = 480, kMapN = kMap * kMap;            // 230400
constexpr int kB = 4;
constexpr int kX1 = 190, kY1 = 240;
constexpr float kDeg = 57.29577951308232f;
constexpr int kRows  = kCFeat * kVR * kVR;                // 170000 voxel columns
constexpr int kTiles = kRows / 16;                        // 10625 (exact)
}

// ---------------------------------------------------------------- zero fill
__global__ void sm_zero(float* __restrict__ p, int n) {
  int i = blockIdx.x * blockDim.x + threadIdx.x;
  int stride = gridDim.x * blockDim.x;
  for (; i < n; i += stride) p[i] = 0.0f;
}

// ---------------------------------------------------------------- pose update
__global__ void sm_pose(const float* __restrict__ pose_obs,
                        const float* __restrict__ poses_last,
                        float* __restrict__ pose_out1,
                        float* __restrict__ pose_out2,
                        float* __restrict__ params) {
  int b = threadIdx.x;
  if (b >= kB) return;
  float dx = pose_obs[b * 3 + 0], dy = pose_obs[b * 3 + 1], dth = pose_obs[b * 3 + 2];
  float px = poses_last[b * 3 + 0], py = poses_last[b * 3 + 1], pt = poses_last[b * 3 + 2];
  float t0 = pt / kDeg;
  float s0 = sinf(t0), c0 = cosf(t0);
  float y_new = py + dx * s0 + dy * c0;
  float x_new = px + dx * c0 - dy * s0;
  float t_new = pt + dth * kDeg;
  t_new = fmodf(t_new - 180.0f, 360.0f) + 180.0f;
  t_new = fmodf(t_new + 180.0f, 360.0f) - 180.0f;
  pose_out1[b * 3 + 0] = x_new; pose_out1[b * 3 + 1] = y_new; pose_out1[b * 3 + 2] = t_new;
  pose_out2[b * 3 + 0] = x_new; pose_out2[b * 3 + 1] = y_new; pose_out2[b * 3 + 2] = t_new;
  const float half = 240.0f;
  float stx = -(x_new * 20.0f - half) / half;   // *100/RES = *20
  float sty = -(y_new * 20.0f - half) / half;
  float stt = (90.0f - t_new) * (3.14159265358979323846f / 180.0f);
  params[b * 4 + 0] = cosf(stt);
  params[b * 4 + 1] = sinf(stt);
  params[b * 4 + 2] = stx;
  params[b * 4 + 3] = sty;
}

// ---------------------------------------------------------------- trilinear splat
__global__ void sm_splat(const float* __restrict__ obs,
                         const float* __restrict__ sh,
                         float* __restrict__ vox, int b) {
  int n = blockIdx.x * blockDim.x + threadIdx.x;
  if (n >= kN) return;
  const float* ob = obs + (size_t)b * kCObs * kN;
  float depth = ob[3 * kN + n];
  int i = n / kFW, j = n - i * kFW;
  // f = 320 / tan(39.5 deg); invf = 1/f
  const float invf = tanf(39.5f * 3.14159265358979323846f / 180.0f) * (1.0f / 320.0f);
  float X = ((float)j - 319.5f) * depth * invf;
  float Z = (239.5f - (float)i) * depth * invf;
  float xn = ((X + 250.0f) * 0.2f - 50.0f) * (1.0f / 50.0f);
  float yn = (depth * 0.2f - 50.0f) * (1.0f / 50.0f);
  float zn = ((Z + 88.0f) * 0.2f - 32.0f) * (1.0f / 40.0f);
  xn = fminf(fmaxf(xn, -1.0f), 1.0f);
  yn = fminf(fmaxf(yn, -1.0f), 1.0f);
  zn = fminf(fmaxf(zn, -1.0f), 1.0f);
  float p0 = xn * 50.0f + 50.0f;
  float p1 = yn * 50.0f + 50.0f;
  float p2 = zn * 40.0f + 40.0f;
  const float* shp = sh + ((size_t)b * kN + n) * 4;
  float shm  = 0.25f * (shp[0] + shp[1] + shp[2] + shp[3]);
  float enh0 = 1.0f + shm;
  float fsem[kNSem];
#pragma unroll
  for (int c = 0; c < kNSem; ++c) fsem[c] = ob[(4 + c) * kN + n] * enh0;
  float f0 = floorf(p0), f1 = floorf(p1), f2 = floorf(p2);
#pragma unroll
  for (int t = 0; t < 8; ++t) {
    float px = f0 + (float)((t >> 2) & 1);
    float py = f1 + (float)((t >> 1) & 1);
    float pz = f2 + (float)(t & 1);
    if (!(px > 0.0f && px < 100.0f && py > 0.0f && py < 100.0f &&
          pz > 0.0f && pz < 80.0f))
      continue;
    float w = (1.0f - fabsf(p0 - px)) * (1.0f - fabsf(p1 - py)) *
              (1.0f - fabsf(p2 - pz));
    if (!(w > 0.0f)) continue;
    int idx = (((int)px) * 100 + (int)py) * 80 + (int)pz;
    float* vp = vox + idx;
    atomicAdd(vp, enh0 * w);
#pragma unroll
    for (int c = 0; c < kNSem; ++c)
      atomicAdd(vp + (size_t)(c + 1) * kVoxN, fsem[c] * w);
  }
}

// ---------------------------------------------------------------- WMMA z-reduction
// D = A(16 cols x 32 z, f16) * B(32 x 16, f16) + C.  B col0 = ones (all-z sum),
// B col1 = agent mask (z in [13,25)).  Three k-chunks cover z = 0..95 (z>=80 zeroed).
__global__ void sm_reduce_wmma(const float* __restrict__ vox,
                               float* __restrict__ av,
                               float* __restrict__ fp_out) {
  int wave = (blockIdx.x * blockDim.x + threadIdx.x) >> 5;
  int lane = threadIdx.x & 31;
  if (wave >= kTiles) return;   // wave-uniform: EXEC stays all-ones around wmma
  int tile = wave;
  int row = tile * 16 + (lane & 15);
  int c   = row / 10000;
  int rem = row - c * 10000;
  int x   = rem / 100;
  int y   = rem - x * 100;
  const float* col = vox + (size_t)((c * 100 + x) * 100 + y) * kHZ;
  int kgroup = (lane >> 4) * 8;    // A: lane half selects K 0-7/16-23 vs 8-15/24-31
  int khalf  = (lane >> 4) * 16;   // B: uses kgroup, same K order as A (0-7/16-23 vs 8-15/24-31)
  int colN   = lane & 15;          // B: lane selects output column N
  v8f acc = {};
#pragma unroll
  for (int k0 = 0; k0 < 96; k0 += 32) {
    int z0 = k0 + kgroup;
    int z1 = k0 + kgroup + 16;
    float4 q0 = (z0     < kHZ) ? *(const float4*)(col + z0)     : make_float4(0.f, 0.f, 0.f, 0.f);
    float4 q1 = (z0 + 4 < kHZ) ? *(const float4*)(col + z0 + 4) : make_float4(0.f, 0.f, 0.f, 0.f);
    float4 q2 = (z1     < kHZ) ? *(const float4*)(col + z1)     : make_float4(0.f, 0.f, 0.f, 0.f);
    float4 q3 = (z1 + 4 < kHZ) ? *(const float4*)(col + z1 + 4) : make_float4(0.f, 0.f, 0.f, 0.f);
    float va[16] = {q0.x, q0.y, q0.z, q0.w, q1.x, q1.y, q1.z, q1.w,
                    q2.x, q2.y, q2.z, q2.w, q3.x, q3.y, q3.z, q3.w};
    v16h a;
#pragma unroll
    for (int e = 0; e < 16; ++e) a[e] = (_Float16)rintf(va[e]);   // jnp.round (RNE)
    v16h bm;
#pragma unroll
    for (int e = 0; e < 16; ++e) {
      float bv = 0.0f;
      if (colN == 0) bv = 1.0f;
      else if (colN == 1) {
        int z = k0 + kgroup + (e < 8 ? e : e + 8);
        bv = (z >= 13 && z < 25) ? 1.0f : 0.0f;
      }
      bm[e] = (_Float16)bv;
    }
    acc = __builtin_amdgcn_wmma_f32_16x16x32_f16(false, a, false, bm,
                                                 (short)0, acc, false, false);
  }
  // C/D layout: VGPR r holds rows M=r (lanes 0-15) / M=r+8 (lanes 16-31), N=lane&15.
  int g = lane & 15;
  if (g < 2) {
    int mhalf = (lane >> 4) * 8;
#pragma unroll
    for (int r = 0; r < 8; ++r) {
      int rr   = tile * 16 + mhalf + r;
      int cc   = rr / 10000;
      int rrem = rr - cc * 10000;
      int xx   = rrem / 100;
      int yy   = rrem - xx * 100;
      float v  = acc[r];
      int base = (kY1 + yy) * kMap + (kX1 + xx);
      if (g == 1) {                       // agent_proj (z in [13,25))
        if (cc == 0) {
          float fp = fminf(fmaxf(v, 0.0f), 1.0f);        // /MAP_T=1
          fp_out[yy * 100 + xx] = fp;
          av[base] = fp;                                  // av channel 0
        } else {
          av[(size_t)(cc + 3) * kMapN + base] =           // av channels 4..19, /CAT_T=5
              fminf(fmaxf(v * 0.2f, 0.0f), 1.0f);
        }
      } else {                            // g == 0: all-z sum
        if (cc == 0)
          av[(size_t)kMapN + base] = fminf(fmaxf(v, 0.0f), 1.0f);  // av channel 1, /EXP_T=1
      }
    }
  }
}

// ---------------------------------------------------------------- bilinear sample
__device__ __forceinline__ float sm_tap(const float* __restrict__ img, int ix, int iy) {
  if ((unsigned)ix >= (unsigned)kMap || (unsigned)iy >= (unsigned)kMap) return 0.0f;
  return img[iy * kMap + ix];
}

__device__ __forceinline__ float sm_bilin(const float* __restrict__ img,
                                          float xg, float yg) {
  float xp = (xg + 1.0f) * 0.5f * 479.0f;
  float yp = (yg + 1.0f) * 0.5f * 479.0f;
  float x0 = floorf(xp), y0 = floorf(yp);
  float fx = xp - x0, fy = yp - y0;
  int ix = (int)x0, iy = (int)y0;
  float v00 = sm_tap(img, ix,     iy);
  float v01 = sm_tap(img, ix + 1, iy);
  float v10 = sm_tap(img, ix,     iy + 1);
  float v11 = sm_tap(img, ix + 1, iy + 1);
  return v00 * (1.0f - fx) * (1.0f - fy) + v01 * fx * (1.0f - fy) +
         v10 * (1.0f - fx) * fy          + v11 * fx * fy;
}

// ---------------------------------------------------------------- rotation sample
__global__ void sm_rot(const float* __restrict__ av, float* __restrict__ rot,
                       const float* __restrict__ params, int b) {
  int idx = blockIdx.x * blockDim.x + threadIdx.x;
  if (idx >= kCObs * kMapN) return;
  int c = idx / kMapN;
  int pix = idx - c * kMapN;
  int h = pix / kMap, w = pix - h * kMap;
  float gx = -1.0f + 2.0f * (float)w / 479.0f;
  float gy = -1.0f + 2.0f * (float)h / 479.0f;
  float ct = params[b * 4 + 0], st = params[b * 4 + 1];
  float xg = ct * gx - st * gy;
  float yg = st * gx + ct * gy;
  rot[idx] = sm_bilin(av + (size_t)c * kMapN, xg, yg);
}

// ---------------------------------------------------------------- translate + max
__global__ void sm_trans(const float* __restrict__ rot,
                         const float* __restrict__ maps_last,
                         float* __restrict__ map_out,
                         const float* __restrict__ params, int b) {
  int idx = blockIdx.x * blockDim.x + threadIdx.x;
  if (idx >= kCObs * kMapN) return;
  int c = idx / kMapN;
  int pix = idx - c * kMapN;
  int h = pix / kMap, w = pix - h * kMap;
  float gx = -1.0f + 2.0f * (float)w / 479.0f;
  float gy = -1.0f + 2.0f * (float)h / 479.0f;
  float xg = gx + params[b * 4 + 2];
  float yg = gy + params[b * 4 + 3];
  float v = sm_bilin(rot + (size_t)c * kMapN, xg, yg);
  size_t o = (size_t)b * kCObs * kMapN + idx;
  map_out[o] = fmaxf(maps_last[o], v);
}

// ---------------------------------------------------------------- launch
extern "C" void kernel_launch(void* const* d_in, const int* in_sizes, int n_in,
                              void* d_out, int out_size, void* d_ws, size_t ws_size,
                              hipStream_t stream) {
  const float* obs        = (const float*)d_in[0];
  const float* pose_obs   = (const float*)d_in[1];
  const float* maps_last  = (const float*)d_in[2];
  const float* poses_last = (const float*)d_in[3];
  const float* sh         = (const float*)d_in[4];

  float* out = (float*)d_out;
  float* fp_out    = out;                                   // (4,1,100,100)
  float* map_out   = out + (size_t)kB * 10000;              // (4,20,480,480)
  float* pose_out1 = map_out + (size_t)kB * kCObs * kMapN;  // (4,3)
  float* pose_out2 = pose_out1 + kB * 3;                    // (4,3)

  float* ws  = (float*)d_ws;
  float* vox = ws;                                          // 17*800000 = 13.6M f32
  float* av  = vox + (size_t)kCFeat * kVoxN;                // 20*230400 = 4.608M f32
  float* rot = av  + (size_t)kCObs * kMapN;                 // 4.608M f32
  float* params = rot + (size_t)kCObs * kMapN;              // 16 f32

  sm_pose<<<1, 32, 0, stream>>>(pose_obs, poses_last, pose_out1, pose_out2, params);

  const int mapThreads = kCObs * kMapN;                     // 4,608,000
  const int mapBlocks  = (mapThreads + 255) / 256;          // 18000
  const int redBlocks  = (kTiles * 32 + 255) / 256;         // 1329

  for (int b = 0; b < kB; ++b) {
    sm_zero<<<4096, 256, 0, stream>>>(vox, kCFeat * kVoxN);
    sm_splat<<<kN / 256, 256, 0, stream>>>(obs, sh, vox, b);
    sm_zero<<<4096, 256, 0, stream>>>(av, kCObs * kMapN);
    sm_reduce_wmma<<<redBlocks, 256, 0, stream>>>(vox, av, fp_out + (size_t)b * 10000);
    sm_rot<<<mapBlocks, 256, 0, stream>>>(av, rot, params, b);
    sm_trans<<<mapBlocks, 256, 0, stream>>>(rot, maps_last, map_out, params, b);
  }
}
